// MultiHeadAttention_19559281066349
// MI455X (gfx1250) — matrix operand order repaired and checked by reading, not compiled, not run on hardware
//
#include <hip/hip_runtime.h>
#include <stdint.h>

// MI455X / gfx1250 causal multi-head attention.  B=4, N=2048, C=1024, H=16, D=64.
// All matrix math via v_wmma_f32_16x16x32_f16; GEMM tiles staged by the
// Tensor Data Mover (tensor_load_to_lds, TENSORcnt) with double buffering.

typedef __attribute__((ext_vector_type(16))) _Float16 v16h;
typedef __attribute__((ext_vector_type(8)))  float    v8f;

constexpr int Bc = 4, NSEQ = 2048, Cc = 1024, Hh = 16, Dd = 64;
constexpr int Mtot = Bc * NSEQ;          // 8192 token rows
constexpr float SCALE = 0.125f;          // D^-0.5

union Frag16 { v16h h; float4 f4[2]; uint32_t u[8]; };

__device__ __forceinline__ v8f wmma_f16(const v16h& a, const v16h& b, const v8f& c) {
  return __builtin_amdgcn_wmma_f32_16x16x32_f16(false, a, false, b, (short)0, c,
                                                false, false);
}

// -------- Tensor Data Mover helpers ----------------------------------------
#if defined(__has_builtin)
#if __has_builtin(__builtin_amdgcn_tensor_load_to_lds)
#define USE_TDM 1
#endif
#endif
#ifndef USE_TDM
#define USE_TDM 0
#endif

#if USE_TDM
typedef unsigned int v4u  __attribute__((ext_vector_type(4)));
typedef int          v4i  __attribute__((ext_vector_type(4)));
typedef int          v8it __attribute__((ext_vector_type(8)));

__device__ __forceinline__ unsigned lds_off(const void* p) {
  // LDS aperture: generic addr[31:0] is the LDS byte offset.
  return (unsigned)(uintptr_t)p;
}

// 2D f16 tile load, global -> LDS, with LDS row padding for bank-conflict-free
// fragment reads.  g1w0 carries data_size/pad_enable/pad_interval bits.
__device__ __forceinline__ void tdm_load_2d(unsigned lds, const void* g,
                                            unsigned g1w0,
                                            unsigned dim0, unsigned dim1,
                                            unsigned tile0, unsigned tile1,
                                            unsigned stride0) {
  uint64_t ga = (uint64_t)(uintptr_t)g;
  v4u g0 = { 1u /*count=1*/, lds, (unsigned)ga,
             (unsigned)(ga >> 32) | 0x80000000u /*type=2 (image)*/ };
  v8it g1 = { (int)g1w0,
              (int)(dim0 << 16),                       // tensor_dim0 lo
              (int)((dim0 >> 16) | (dim1 << 16)),      // dim0 hi | dim1 lo
              (int)((dim1 >> 16) | (tile0 << 16)),     // dim1 hi | tile_dim0
              (int)tile1,                              // tile_dim1 (tile_dim2=0)
              (int)stride0,                             // tensor_dim0_stride lo
              0, 0 };
  v4i z4 = { 0, 0, 0, 0 };
#if __clang_major__ >= 23
  v8it z8 = { 0, 0, 0, 0, 0, 0, 0, 0 };
  __builtin_amdgcn_tensor_load_to_lds(g0, g1, z4, z4, z8, 0);
#else
  __builtin_amdgcn_tensor_load_to_lds(g0, g1, z4, z4, 0);
#endif
}
// g1 word0 fields: data_size=1 (2B) at bit16, pad_enable bit20, pad_interval bits24:22
constexpr unsigned G1_A = 0x00D10000u;  // pad after 16 DW (64B row) -> LDA = 34 f16
constexpr unsigned G1_B = 0x01510000u;  // pad after 64 DW (256B row) -> LDW = 130 f16
#endif

// ---------------------------------------------------------------------------
// f32 -> f16 elementwise conversion (activations).
// ---------------------------------------------------------------------------
__global__ __launch_bounds__(256) void cvt_kernel(const float* __restrict__ src,
                                                  _Float16* __restrict__ dst, int n) {
  int i = (blockIdx.x * 256 + threadIdx.x) * 4;
  if (i < n) {
    float4 v = *reinterpret_cast<const float4*>(src + i);
    dst[i + 0] = (_Float16)v.x; dst[i + 1] = (_Float16)v.y;
    dst[i + 2] = (_Float16)v.z; dst[i + 3] = (_Float16)v.w;
  }
}

// W[C,C] f32 -> Wt[C,C] f16 transposed (Wt[k][n] = W[n][k]), tiled via LDS.
__global__ __launch_bounds__(256) void transpose_cvt_kernel(const float* __restrict__ W,
                                                            _Float16* __restrict__ Wt) {
  __shared__ float t[32][33];
  const int n0 = blockIdx.x * 32, k0 = blockIdx.y * 32;
#pragma unroll
  for (int i = 0; i < 4; ++i) {
    int r = threadIdx.y + 8 * i;
    t[r][threadIdx.x] = W[(size_t)(n0 + r) * Cc + k0 + threadIdx.x];
  }
  __syncthreads();
#pragma unroll
  for (int i = 0; i < 4; ++i) {
    int r = threadIdx.y + 8 * i;
    Wt[(size_t)(k0 + r) * Cc + n0 + threadIdx.x] = (_Float16)t[threadIdx.x][r];
  }
}

// ---------------------------------------------------------------------------
// Tiled GEMM: Y[M,Cc] = X[M,Cc](f16) @ Wt[Cc,Cc](f16, K-major) + bias.
// Block: 256 threads = 8 waves, block tile 128x128, wave tile 32x64, K-step 32.
// TDM double-buffered LDS staging; padded strides LDA=34, LDW=130 f16.
// MODE 0: f32 [M,Cc]   MODE 1: f16 [B,H,N,D]   MODE 2: f16 [B,H,D,N]
// ---------------------------------------------------------------------------
template<int MODE>
__global__ __launch_bounds__(256) void gemm_kernel(const _Float16* __restrict__ X,
                                                   const _Float16* __restrict__ Wt,
                                                   const float* __restrict__ bias,
                                                   float* __restrict__ outF,
                                                   _Float16* __restrict__ outH) {
  constexpr int K     = Cc;   // 1024
  constexpr int KSTEP = 32;
  constexpr int NIT   = K / KSTEP;
  constexpr int LDA   = 34;   // f16 stride (32 + 1 DW pad) -> distinct banks
  constexpr int LDW   = 130;  // f16 stride (128 + 1 DW pad)
  __shared__ _Float16 lA[2][128 * LDA];   // 2 x 8704 B
  __shared__ _Float16 lW[2][32 * LDW];    // 2 x 8320 B

  const int tid  = threadIdx.x;
  const int wave = tid >> 5, lane = tid & 31;
  const int half = lane >> 4, ml = lane & 15;
  const int m0 = blockIdx.y * 128;
  const int n0 = blockIdx.x * 128;
  const int wr = (wave & 3) * 32;   // wave rows within block tile
  const int wc = (wave >> 2) * 64;  // wave cols within block tile

  v8f acc[2][4] = {};

#if USE_TDM
  if (tid < 32) {  // one wave drives the TDM (EXEC-independent, wave-level op)
    tdm_load_2d(lds_off(lA[0]), X + (size_t)m0 * K, G1_A, K, Mtot, KSTEP, 128, K);
    tdm_load_2d(lds_off(lW[0]), Wt + n0, G1_B, Cc, Cc, 128, KSTEP, Cc);
  }
#endif

  for (int it = 0; it < NIT; ++it) {
    const int k0 = it * KSTEP;
#if USE_TDM
    if (tid < 32) {
      if (it + 1 < NIT) {  // prefetch next K tile into the other buffer
        const int kn = k0 + KSTEP;
        tdm_load_2d(lds_off(lA[(it + 1) & 1]), X + (size_t)m0 * K + kn,
                    G1_A, K, Mtot, KSTEP, 128, K);
        tdm_load_2d(lds_off(lW[(it + 1) & 1]), Wt + (size_t)kn * Cc + n0,
                    G1_B, Cc, Cc, 128, KSTEP, Cc);
        // TENSORcnt is in-order per wave: <=2 outstanding => current tile done.
        __builtin_amdgcn_s_wait_tensorcnt(2);
      } else {
        __builtin_amdgcn_s_wait_tensorcnt(0);
      }
    }
    __syncthreads();
    const uint32_t* cA = (const uint32_t*)lA[it & 1];
    const uint32_t* cW = (const uint32_t*)lW[it & 1];
#else
    // Fallback: cooperative staging, single buffer.
#pragma unroll
    for (int i = 0; i < 2; ++i) {
      int slot = tid + 256 * i;                  // 512 chunks of 8 f16
      int r = slot >> 2, c8 = (slot & 3) * 8;
      float4 v = *reinterpret_cast<const float4*>(X + (size_t)(m0 + r) * K + k0 + c8);
      *reinterpret_cast<float2*>(&lA[0][r * LDA + c8])     = make_float2(v.x, v.y);
      *reinterpret_cast<float2*>(&lA[0][r * LDA + c8 + 4]) = make_float2(v.z, v.w);
    }
#pragma unroll
    for (int i = 0; i < 2; ++i) {
      int slot = tid + 256 * i;                  // 512 chunks of 8 f16
      int r = slot >> 4, c8 = (slot & 15) * 8;
      float4 v = *reinterpret_cast<const float4*>(Wt + (size_t)(k0 + r) * Cc + n0 + c8);
      *reinterpret_cast<float2*>(&lW[0][r * LDW + c8])     = make_float2(v.x, v.y);
      *reinterpret_cast<float2*>(&lW[0][r * LDW + c8 + 4]) = make_float2(v.z, v.w);
    }
    __syncthreads();
    const uint32_t* cA = (const uint32_t*)lA[0];
    const uint32_t* cW = (const uint32_t*)lW[0];
#endif

    // A fragments (16x32): lane(0-15)=row, K pairs split by lane-half.
    Frag16 a[2], b[4];
#pragma unroll
    for (int t = 0; t < 2; ++t) {
      int base = (wr + 16 * t + ml) * (LDA / 2) + half * 4;
#pragma unroll
      for (int j = 0; j < 4; ++j) {
        a[t].u[j]     = cA[base + j];       // K = half*8 + 2j
        a[t].u[4 + j] = cA[base + 8 + j];   // K = half*8 + 16 + 2j
      }
    }
    // B fragments (32x16): lane(0-15)=col, K split by lane-half (same order as A).
#pragma unroll
    for (int t = 0; t < 4; ++t) {
      const _Float16* hW = (const _Float16*)cW + wc + 16 * t + ml;
#pragma unroll
      for (int s = 0; s < 16; ++s) b[t].h[s] = hW[((s & 7) + 8 * half + ((s >> 3) << 4)) * LDW];
    }
#pragma unroll
    for (int i = 0; i < 2; ++i)
#pragma unroll
      for (int j = 0; j < 4; ++j)
        acc[i][j] = wmma_f16(a[i].h, b[j].h, acc[i][j]);
    __syncthreads();
  }

  // Epilogue. C/D layout: VGPR j -> M = j + 8*half, N = lane&15.
#pragma unroll
  for (int i = 0; i < 2; ++i) {
#pragma unroll
    for (int t = 0; t < 4; ++t) {
      int col = n0 + wc + 16 * t + ml;
      float bval = bias[col];
#pragma unroll
      for (int j = 0; j < 8; ++j) {
        int row = m0 + wr + 16 * i + j + 8 * half;
        float y = acc[i][t][j] + bval;
        if (MODE == 0) {
          outF[(size_t)row * Cc + col] = y;
        } else {
          int bi = row >> 11, n = row & (NSEQ - 1);
          int h = col >> 6, d = col & (Dd - 1);
          if (MODE == 1)
            outH[(((size_t)(bi * Hh + h)) * NSEQ + n) * Dd + d] = (_Float16)y;
          else
            outH[(((size_t)(bi * Hh + h)) * Dd + d) * NSEQ + n] = (_Float16)y;
        }
      }
    }
  }
}

// ---------------------------------------------------------------------------
// Flash attention: one wave per 16 query rows, key tiles of 32, online softmax.
// Q,V: f16 [B,H,N,D]; Kt: f16 [B,H,D,N]; output f16 token-major [B,N,C].
// ---------------------------------------------------------------------------
__global__ __launch_bounds__(128) void attn_kernel(const _Float16* __restrict__ Q,
                                                   const _Float16* __restrict__ Kt,
                                                   const _Float16* __restrict__ V,
                                                   _Float16* __restrict__ att) {
  constexpr int PST = 36;
  __shared__ _Float16 pbuf[4][16 * PST];

  const int tid  = threadIdx.x;
  const int wave = tid >> 5, lane = tid & 31;
  const int half = lane >> 4, ml = lane & 15;
  const int bh = blockIdx.y;
  const int b = bh / Hh, h = bh % Hh;
  const int q0 = (blockIdx.x * 4 + wave) * 16;

  const _Float16* qb = Q  + (size_t)bh * NSEQ * Dd;
  const _Float16* kb = Kt + (size_t)bh * Dd * NSEQ;
  const _Float16* vb = V  + (size_t)bh * NSEQ * Dd;

  Frag16 aq[2];
  {
    int row = q0 + ml, kbo = half * 8;
#pragma unroll
    for (int c = 0; c < 2; ++c) {
      const _Float16* p = qb + (size_t)row * Dd + c * 32 + kbo;
      aq[c].f4[0] = *reinterpret_cast<const float4*>(p);
      aq[c].f4[1] = *reinterpret_cast<const float4*>(p + 16);
      aq[c].h = aq[c].h * (_Float16)SCALE;
    }
  }

  float mrow[8], lrow[8];
  v8f accO[4] = {};
#pragma unroll
  for (int j = 0; j < 8; ++j) { mrow[j] = -1e30f; lrow[j] = 0.0f; }

  _Float16* pb = pbuf[wave];
  const uint32_t* pbu = (const uint32_t*)pb;

  const int kend   = q0 + 16;
  const int ntiles = (kend + 31) >> 5;
  for (int t = 0; t < ntiles; ++t) {
    const int j0 = t << 5;
    v8f s0 = {}, s1 = {};
#pragma unroll
    for (int c = 0; c < 2; ++c) {
      const _Float16* kp = kb + (size_t)(c * 32) * NSEQ + j0 + ml;
      Frag16 kf0, kf1;
#pragma unroll
      for (int s = 0; s < 16; ++s) {
        size_t o = (size_t)((s & 7) + 8 * half + ((s >> 3) << 4)) * NSEQ;
        kf0.h[s] = kp[o];
        kf1.h[s] = kp[o + 16];
      }
      s0 = wmma_f16(aq[c].h, kf0.h, s0);
      s1 = wmma_f16(aq[c].h, kf1.h, s1);
    }
    if (j0 + 32 > kend) {  // causal mask on the diagonal tile
      const int col0 = j0 + ml;
#pragma unroll
      for (int j = 0; j < 8; ++j) {
        int row = q0 + j + 8 * half;
        if (col0 > row)      s0[j] = -1e30f;
        if (col0 + 16 > row) s1[j] = -1e30f;
      }
    }
#pragma unroll
    for (int j = 0; j < 8; ++j) {
      float a0 = s0[j], a1 = s1[j];
      float mx = fmaxf(a0, a1);
#pragma unroll
      for (int off = 1; off < 16; off <<= 1)
        mx = fmaxf(mx, __shfl_xor(mx, off, 32));
      float mnew = fmaxf(mrow[j], mx);
      float corr = __expf(mrow[j] - mnew);
      float p0 = __expf(a0 - mnew);
      float p1 = __expf(a1 - mnew);
      float rs = p0 + p1;
#pragma unroll
      for (int off = 1; off < 16; off <<= 1)
        rs += __shfl_xor(rs, off, 32);
      lrow[j] = lrow[j] * corr + rs;
      mrow[j] = mnew;
#pragma unroll
      for (int c = 0; c < 4; ++c) accO[c][j] *= corr;
      int prow = j + 8 * half;
      pb[prow * PST + ml]      = (_Float16)p0;
      pb[prow * PST + 16 + ml] = (_Float16)p1;
    }
    asm volatile("" ::: "memory");
    asm volatile("s_wait_dscnt 0" ::: "memory");   // same-wave LDS round-trip
    Frag16 pa;
    {
      int base = ml * (PST / 2) + half * 4;
#pragma unroll
      for (int j = 0; j < 4; ++j) {
        pa.u[j]     = pbu[base + j];
        pa.u[4 + j] = pbu[base + 8 + j];
      }
    }
    const _Float16* vp = vb + (size_t)j0 * Dd + ml;
#pragma unroll
    for (int c = 0; c < 4; ++c) {
      Frag16 bv;
#pragma unroll
      for (int s = 0; s < 16; ++s)
        bv.h[s] = vp[((s & 7) + 8 * half + ((s >> 3) << 4)) * Dd + c * 16];
      accO[c] = wmma_f16(pa.h, bv.h, accO[c]);
    }
  }

#pragma unroll
  for (int j = 0; j < 8; ++j) {
    float inv = 1.0f / lrow[j];
    int row = q0 + j + 8 * half;
    size_t tok = (size_t)b * NSEQ + row;
#pragma unroll
    for (int c = 0; c < 4; ++c)
      att[tok * Cc + h * Dd + c * 16 + ml] = (_Float16)(accO[c][j] * inv);
  }
}

// ---------------------------------------------------------------------------
extern "C" void kernel_launch(void* const* d_in, const int* in_sizes, int n_in,
                              void* d_out, int out_size, void* d_ws, size_t ws_size,
                              hipStream_t stream) {
  (void)in_sizes; (void)n_in; (void)out_size; (void)ws_size;
  const float* query = (const float*)d_in[0];
  const float* key   = (const float*)d_in[1];
  const float* value = (const float*)d_in[2];
  // d_in[3]: causal bool mask, applied analytically in attn_kernel.
  const float* Wq = (const float*)d_in[4];
  const float* bq = (const float*)d_in[5];
  const float* Wk = (const float*)d_in[6];
  const float* bk = (const float*)d_in[7];
  const float* Wv = (const float*)d_in[8];
  const float* bv = (const float*)d_in[9];
  const float* Wo = (const float*)d_in[10];
  const float* bo = (const float*)d_in[11];

  // Workspace (f16 elements): 3x act(8M) + 4x Wt(1M) + q/kT/v(8M each) + att(8M)
  const size_t nAct = (size_t)Mtot * Cc;       // 8M
  const size_t nW   = (size_t)Cc * Cc;         // 1M
  _Float16* xq  = (_Float16*)d_ws;
  _Float16* xk  = xq + nAct;
  _Float16* xv  = xk + nAct;
  _Float16* wtq = xv + nAct;
  _Float16* wtk = wtq + nW;
  _Float16* wtv = wtk + nW;
  _Float16* wto = wtv + nW;
  _Float16* qf  = wto + nW;
  _Float16* ktf = qf + nAct;
  _Float16* vf  = ktf + nAct;
  _Float16* att = vf + nAct;

  // 1) dtype conversion / weight transposition (bandwidth-trivial).
  const int nconv = (int)nAct;
  dim3 cblk(256), cgrid(nconv / (256 * 4));
  cvt_kernel<<<cgrid, cblk, 0, stream>>>(query, xq, nconv);
  cvt_kernel<<<cgrid, cblk, 0, stream>>>(key,   xk, nconv);
  cvt_kernel<<<cgrid, cblk, 0, stream>>>(value, xv, nconv);
  dim3 tblk(32, 8), tgrid(Cc / 32, Cc / 32);
  transpose_cvt_kernel<<<tgrid, tblk, 0, stream>>>(Wq, wtq);
  transpose_cvt_kernel<<<tgrid, tblk, 0, stream>>>(Wk, wtk);
  transpose_cvt_kernel<<<tgrid, tblk, 0, stream>>>(Wv, wtv);
  transpose_cvt_kernel<<<tgrid, tblk, 0, stream>>>(Wo, wto);

  // 2) QKV projections (WMMA + TDM staging).
  dim3 gblk(256), ggrid(Cc / 128, Mtot / 128);
  gemm_kernel<1><<<ggrid, gblk, 0, stream>>>(xq, wtq, bq, nullptr, qf);
  gemm_kernel<2><<<ggrid, gblk, 0, stream>>>(xk, wtk, bk, nullptr, ktf);
  gemm_kernel<1><<<ggrid, gblk, 0, stream>>>(xv, wtv, bv, nullptr, vf);

  // 3) Flash attention.
  dim3 ablk(128), agrid(NSEQ / 64, Bc * Hh);
  attn_kernel<<<agrid, ablk, 0, stream>>>(qf, ktf, vf, att);

  // 4) Output projection straight into d_out (f32).
  gemm_kernel<0><<<ggrid, gblk, 0, stream>>>(att, wto, bo, (float*)d_out, nullptr);
}
